// elm_ae_80522046866068
// MI455X (gfx1250) — matrix operand order repaired and checked by reading, not compiled, not run on hardware
//
#include <hip/hip_runtime.h>
#include <hip/hip_bf16.h>
#include <math.h>

typedef __attribute__((ext_vector_type(16))) _Float16 v16h;
typedef __attribute__((ext_vector_type(8)))  float    v8f;

#define HIDDEN 512
#define MDIM   28
#define HS     514   // Hs row stride in f16 elems: 514*2B=1028B -> bank shift 1 dword/row, conflict-free
#define XS     33    // f32 scratch row stride: 33 dwords -> conflict-free column access

__device__ __forceinline__ v8f wmma_f16(v16h a, v16h b, v8f c) {
  // (neg_a, A, neg_b, B, c_mod, C, reuse_a, reuse_b)
  return __builtin_amdgcn_wmma_f32_16x16x32_f16(false, a, false, b, (short)0, c, false, false);
}

// A-frag K-offset for 16x32 f16 A tile, VGPR r, lane-half 'half'
__device__ __forceinline__ int ka_off(int r, int half) {
  return 2 * (r & 3) + 16 * (r >> 2) + 8 * half;
}
// B-frag K-offset for 32x16 f16 B tile
__device__ __forceinline__ int kb_off(int r, int half) {
  return 2 * (r & 3) + 16 * (r >> 2) + 8 * half;
}

__global__ __launch_bounds__(256)
void elm_ae_kernel(const float* __restrict__ xg, const float* __restrict__ Wg,
                   const float* __restrict__ bg, float* __restrict__ Xout,
                   float* __restrict__ Bout, int N)
{
  __shared__ _Float16 Hs[32 * HS];   // H (f16), rows j=0..31 (x-row), cols h=0..511
  __shared__ float xs[32 * XS];      // x padded to 32x32 (zero pad)
  __shared__ float gs[32 * XS];      // G, factored in place (Cholesky L)
  __shared__ float gso[32 * XS];     // untouched copy of G for X = G*Y
  __shared__ float ys[32 * XS];      // Y = G^{-1} x (pad rows zero)
  __shared__ float invd[MDIM];       // 1 / diag(L)

  const int tid  = threadIdx.x;
  const int lane = tid & 31;
  const int wave = tid >> 5;
  const int ln16 = lane & 15;
  const int half = lane >> 4;
  const int sample = blockIdx.x;
  const float* xsamp = xg + (size_t)sample * (MDIM * MDIM);

  // ---- init LDS: zero x-pad and Y ----
  for (int i = tid; i < 32 * XS; i += 256) { xs[i] = 0.0f; ys[i] = 0.0f; }
  __syncthreads();
  for (int i = tid; i < MDIM * MDIM; i += 256) {
    int r = i / MDIM, c = i - r * MDIM;
    xs[r * XS + c] = xsamp[i];
  }
  __syncthreads();

  // ---- Phase 1: H = sigmoid(x @ W^T + b), K=28 (one K=32 WMMA), 2x32 tiles / 8 waves ----
  #pragma unroll
  for (int t = 0; t < 8; ++t) {
    int p = wave * 8 + t;
    int mt = p & 1, nt = p >> 1;            // mt: x-row tile, nt: hidden tile
    int m = mt * 16 + ln16;
    int h = nt * 16 + ln16;
    v16h a, b;
    #pragma unroll
    for (int r = 0; r < 8; ++r) {
      int ka = ka_off(r, half);
      a[2*r]   = (_Float16)xs[m * XS + ka];
      a[2*r+1] = (_Float16)xs[m * XS + ka + 1];
      int kb = kb_off(r, half);             // W^T[kb][h] = W[h][kb]; zero-pad kb>=28
      float w0 = 0.0f, w1 = 0.0f;
      if (kb < MDIM) { w0 = Wg[h * MDIM + kb]; w1 = Wg[h * MDIM + kb + 1]; }
      b[2*r]   = (_Float16)w0;
      b[2*r+1] = (_Float16)w1;
    }
    v8f c = {};
    c = wmma_f16(a, b, c);
    float bias = bg[h];
    #pragma unroll
    for (int r = 0; r < 8; ++r) {
      int row = mt * 16 + r + 8 * half;
      float v = c[r] + bias;
      // sigmoid with fast v_rcp_f32 instead of IEEE div chain
      float s = __builtin_amdgcn_rcpf(1.0f + __expf(-v));
      Hs[row * HS + h] = (_Float16)s;
    }
  }
  __syncthreads();

  // ---- Phase 2: G = H @ H^T, 2x2 tiles on waves 0..3, K=512 (16 WMMA steps) ----
  if (wave < 4) {
    int mt = wave & 1, nt = wave >> 1;
    int m = mt * 16 + ln16;
    int n = nt * 16 + ln16;
    v8f c = {};
    #pragma unroll
    for (int kk = 0; kk < HIDDEN / 32; ++kk) {
      int k0 = kk * 32;
      v16h a, b;
      #pragma unroll
      for (int r = 0; r < 8; ++r) {
        int ka = k0 + ka_off(r, half);
        a[2*r]   = Hs[m * HS + ka];
        a[2*r+1] = Hs[m * HS + ka + 1];
        int kb = k0 + kb_off(r, half);      // H^T[kb][n] = H[n][kb]
        b[2*r]   = Hs[n * HS + kb];
        b[2*r+1] = Hs[n * HS + kb + 1];
      }
      c = wmma_f16(a, b, c);
    }
    #pragma unroll
    for (int r = 0; r < 8; ++r) {
      int row = mt * 16 + r + 8 * half;
      int col = nt * 16 + ln16;
      gs[row * XS + col]  = c[r];
      gso[row * XS + col] = c[r];
    }
  }
  __syncthreads();

  // ---- Phase 3: Cholesky of G (28x28, in place -> L) ----
  // hoist the (i,j) tile mapping for the rank-1 update out of the k loop
  int ui[4], uj[4];
  #pragma unroll
  for (int q = 0; q < 4; ++q) {
    int idx = tid + q * 256;
    int valid = (idx < MDIM * MDIM);
    int i = valid ? (idx / MDIM) : 0;
    int j = valid ? (idx - i * MDIM) : 0;
    ui[q] = valid ? i : -1;
    uj[q] = j;
  }
  for (int k = 0; k < MDIM; ++k) {
    float dk = gs[k * XS + k];
    float rl = __builtin_amdgcn_rsqf(dk);   // 1/sqrt(dk) == 1/L[k][k]
    float lkk = dk * rl;
    __syncthreads();
    if (tid == k)                    gs[k * XS + k] = lkk;
    else if (tid > k && tid < MDIM)  gs[tid * XS + k] *= rl;
    __syncthreads();
    #pragma unroll
    for (int q = 0; q < 4; ++q) {
      int i = ui[q], j = uj[q];
      if (i > k && j > k && j <= i)
        gs[i * XS + j] -= gs[i * XS + k] * gs[j * XS + k];
    }
    __syncthreads();
  }
  // reciprocal diagonal for the solves
  if (tid < MDIM) invd[tid] = __builtin_amdgcn_rcpf(gs[tid * XS + tid]);
  __syncthreads();

  // ---- Phase 4: solve G Y = x (one thread per RHS column) ----
  if (tid < MDIM) {
    const int c = tid;
    float z[MDIM];
    #pragma unroll
    for (int i = 0; i < MDIM; ++i) {        // forward: L z = x[:,c]
      float s = xs[i * XS + c];
      #pragma unroll
      for (int j = 0; j < i; ++j) s -= gs[i * XS + j] * z[j];
      z[i] = s * invd[i];
    }
    #pragma unroll
    for (int i = MDIM - 1; i >= 0; --i) {   // backward: L^T y = z
      float s = z[i];
      #pragma unroll
      for (int j = i + 1; j < MDIM; ++j) s -= gs[j * XS + i] * z[j];
      z[i] = s * invd[i];
    }
    #pragma unroll
    for (int i = 0; i < MDIM; ++i) ys[i * XS + c] = z[i];
  }
  __syncthreads();

  // ---- Phase 5: B = H^T @ Y  (512x28 output), 32x2 tiles / 8 waves, K=32 ----
  // B is 234 MB write-once output: stream it with non-temporal stores (keep L2 for W/x).
  float* Bsamp = Bout + (size_t)sample * (HIDDEN * MDIM);
  #pragma unroll
  for (int t = 0; t < 8; ++t) {
    int p = wave * 8 + t;
    int mt = p >> 1, nt = p & 1;            // mt: hidden tile, nt: col tile
    int m = mt * 16 + ln16;                 // hidden index
    int n = nt * 16 + ln16;                 // output col
    v16h a, b;
    #pragma unroll
    for (int r = 0; r < 8; ++r) {
      int ka = ka_off(r, half);             // H^T[m][ka] = H[ka][m]
      a[2*r]   = Hs[ka * HS + m];
      a[2*r+1] = Hs[(ka + 1) * HS + m];
      int kb = kb_off(r, half);             // Y rows >=28 are zero -> pad safe
      b[2*r]   = (_Float16)ys[kb * XS + n];
      b[2*r+1] = (_Float16)ys[(kb + 1) * XS + n];
    }
    v8f c = {};
    c = wmma_f16(a, b, c);
    if (n < MDIM) {
      #pragma unroll
      for (int r = 0; r < 8; ++r) {
        int row = mt * 16 + r + 8 * half;
        __builtin_nontemporal_store(c[r], &Bsamp[row * MDIM + n]);
      }
    }
  }

  // ---- Phase 6: X = G @ Y (== H @ B), 2x2 tiles on waves 0..3, K=32 ----
  float* Xsamp = Xout + (size_t)sample * (MDIM * MDIM);
  if (wave < 4) {
    int mt = wave & 1, nt = wave >> 1;
    int m = mt * 16 + ln16;
    int n = nt * 16 + ln16;
    v16h a, b;
    #pragma unroll
    for (int r = 0; r < 8; ++r) {
      int ka = ka_off(r, half);             // G pad cols * zero Y pad rows -> safe
      a[2*r]   = (_Float16)gso[m * XS + ka];
      a[2*r+1] = (_Float16)gso[m * XS + ka + 1];
      int kb = kb_off(r, half);
      b[2*r]   = (_Float16)ys[kb * XS + n];
      b[2*r+1] = (_Float16)ys[(kb + 1) * XS + n];
    }
    v8f c = {};
    c = wmma_f16(a, b, c);
    #pragma unroll
    for (int r = 0; r < 8; ++r) {
      int row = mt * 16 + r + 8 * half;
      if (row < MDIM && n < MDIM)
        __builtin_nontemporal_store(c[r], &Xsamp[row * MDIM + n]);
    }
  }
}

extern "C" void kernel_launch(void* const* d_in, const int* in_sizes, int n_in,
                              void* d_out, int out_size, void* d_ws, size_t ws_size,
                              hipStream_t stream) {
  const float* x = (const float*)d_in[0];   // [N,1,28,28] f32
  const float* W = (const float*)d_in[1];   // [512,28]    f32
  const float* b = (const float*)d_in[2];   // [512]       f32
  int N = in_sizes[0] / (MDIM * MDIM);
  float* Xout = (float*)d_out;                       // [N,1,28,28]
  float* Bout = Xout + (size_t)N * (MDIM * MDIM);    // [N,1,512,28]
  elm_ae_kernel<<<dim3(N), dim3(256), 0, stream>>>(x, W, b, Xout, Bout, N);
}
